// RegionProposalNetwork_51582557225592
// MI455X (gfx1250) — matrix operand order repaired and checked by reading, not compiled, not run on hardware
//
#include <hip/hip_runtime.h>

// ---------------------------------------------------------------------------
// RPN (Faster R-CNN) for MI455X / gfx1250, wave32.
// pack(x->HWC f16, W1->rs-major f16, head W->f16) -> conv3x3 implicit GEMM
// (software-pipelined, 64x64 block tile, 2x v_wmma_f32_16x16x32_f16 per wave
// per K-step) -> 1x1 heads (WMMA, direct b128 fragment loads) -> fg scores ->
// anchors/loc2bbox/clip -> bitonic top-K -> bitmask NMS -> compacted rois.
// ---------------------------------------------------------------------------

typedef __attribute__((ext_vector_type(16))) _Float16 v16h;
typedef __attribute__((ext_vector_type(8)))  _Float16 v8h;
typedef __attribute__((ext_vector_type(8)))  float    v8f;

#define HHs 50
#define WWs 50
#define PIX 2500
#define CIN 512
#define COUT 512
#define KTOT 4608           // 9*512, rs-major: k = rs*512 + ci
#define NANCH 22500
#define PPRE 12000
#define NPOST 2000
#define NSORT 32768
#define NBW 188
#define NEG_INF (-__builtin_inff())

// d_out (floats): locs[90000] | scores[45000] | rois[8000] | roiIdx[2000] | anchor[90000]
#define OUT_LOCS   0
#define OUT_SCORES 90000
#define OUT_ROIS   135000
#define OUT_RIDX   143000
#define OUT_ANCH   145000

// workspace byte offsets (256B aligned)
#define WS_XH     ((size_t)0)           // 2500*512 f16 = 2,560,000
#define WS_WPACK  ((size_t)2560000)     // 512*4608 f16 = 4,718,592
#define WS_H16    ((size_t)7278592)     // 2560*512 f16 = 2,621,440
#define WS_WHEAD  ((size_t)9900032)     // 64*512 f16   = 65,536
#define WS_ROI    ((size_t)9965568)     // 22500*4 f32  (pad 360,192)
#define WS_SCOREM ((size_t)10325760)    // 22500 f32    (pad 90,112)
#define WS_KEY    ((size_t)10415872)    // 32768 f32
#define WS_IDX    ((size_t)10546944)    // 32768 i32
#define WS_BOXS   ((size_t)10678016)    // 12000*4 f32  (pad 192,256)
#define WS_VALS   ((size_t)10870272)    // 12000 i32    (pad 48,128)
#define WS_MASK   ((size_t)10918400)    // 12000*188 u64 = 18,048,000

// ---------------------------------------------------------------------------
// Pack kernels (one-time, tiny vs 23.3 TB/s HBM)
// ---------------------------------------------------------------------------
__global__ void pack_x_hwc(const float* __restrict__ x, _Float16* __restrict__ xh)
{
    __shared__ float tile[32][33];
    const int pBase = blockIdx.x * 32, cBase = blockIdx.y * 32;
    const int tx = threadIdx.x, ty = threadIdx.y;           // 32 x 8
    #pragma unroll
    for (int i = 0; i < 4; ++i) {
        int ci = cBase + ty + i * 8;
        int p  = pBase + tx;
        tile[ty + i * 8][tx] = (p < PIX) ? x[(size_t)ci * PIX + p] : 0.f;
    }
    __syncthreads();
    #pragma unroll
    for (int i = 0; i < 4; ++i) {
        int p  = pBase + ty + i * 8;
        int ci = cBase + tx;
        if (p < PIX) xh[(size_t)p * CIN + ci] = (_Float16)tile[tx][ty + i * 8];
    }
}

// W1 [co][ci][3][3] f32 -> wp [co][rs*512+ci] f16 (rs-major K)
__global__ void pack_w1(const float* __restrict__ W1, _Float16* __restrict__ wp)
{
    int i = blockIdx.x * 256 + threadIdx.x;
    if (i >= COUT * KTOT) return;
    int co = i / KTOT, k = i - co * KTOT;
    int rs = k >> 9, ci = k & 511;
    wp[i] = (_Float16)W1[(size_t)co * KTOT + ci * 9 + rs];
}

// head weights: wh [64][512] f16 (o<36: Wl, o<54: Ws, else 0)
__global__ void pack_whead(const float* __restrict__ Wl, const float* __restrict__ Wsc,
                           _Float16* __restrict__ wh)
{
    int i = blockIdx.x * 256 + threadIdx.x;
    if (i >= 64 * CIN) return;
    int o = i >> 9, k = i & 511;
    float w = 0.f;
    if (o < 36)      w = Wl[(size_t)o * CIN + k];
    else if (o < 54) w = Wsc[(size_t)(o - 36) * CIN + k];
    wh[i] = (_Float16)w;
}

// ---------------------------------------------------------------------------
// Kernel 1: 3x3 conv + bias + ReLU, implicit GEMM with WMMA.
// Block 256 thr (8 waves), tile 64 px x 64 cout, waves 4(M) x 2(N), each wave
// two 16x16 tiles sharing one A fragment. Software-pipelined: next K-step's
// global tiles are loaded into registers while the current step computes.
// ---------------------------------------------------------------------------
__device__ __forceinline__ v8h load_a_slice(const _Float16* __restrict__ xh,
                                            int ap, int aoy, int aox, int akk, int k0)
{
    const int rs  = k0 >> 9;              // block-uniform 0..8 (SALU)
    const int cib = k0 & 511;
    const int rr  = rs / 3;
    const int iy  = aoy + rr - 1;
    const int ix  = aox + (rs - rr * 3) - 1;
    v8h r = {};
    if (ap < PIX && (unsigned)iy < (unsigned)HHs && (unsigned)ix < (unsigned)WWs)
        r = *(const v8h*)(xh + (size_t)(iy * WWs + ix) * CIN + cib + akk);
    return r;
}

__global__ void conv3x3_wmma(const _Float16* __restrict__ xh, const _Float16* __restrict__ wp,
                             const float* __restrict__ b1, _Float16* __restrict__ h16)
{
    __shared__ __align__(16) _Float16 As[64 * 32];   // [px_local][k]
    __shared__ __align__(16) _Float16 Bs[64 * 32];   // [co_local][k]
    const int tid  = threadIdx.x;
    const int lane = tid & 31;
    const int wave = tid >> 5;
    const int wm = wave & 3, wn = wave >> 2;
    const int pxBase = blockIdx.y * 64;
    const int coBase = blockIdx.x * 64;

    // A staging geometry: one v8h (16B) per thread
    const int apl = tid >> 2;            // 0..63 local pixel
    const int akk = (tid & 3) * 8;       // 0,8,16,24 within 32-k slice
    const int ap  = pxBase + apl;
    const int aoy = ap / WWs, aox = ap - aoy * WWs;
    // B staging geometry: one v8h (16B) per thread (64 rows x 32 k)
    const int bco = tid >> 2;            // 0..63
    const int bkk = (tid & 3) * 8;       // 0,8,16,24
    const _Float16* wrow = wp + (size_t)(coBase + bco) * KTOT;

    // fragment geometry (contiguous k order, same in A and B)
    const int arow  = wm * 16 + (lane & 15);
    const int akb   = (lane >> 4) * 16;  // A: lanes 0-15 K 0-15, 16-31 K 16-31
    const int bcol0 = wn * 32 + (lane & 15);
    const int bcol1 = bcol0 + 16;
    const int bkb   = (lane >> 4) * 16;  // B: lanes 0-15 K 0-15, 16-31 K 16-31

    v8f acc0 = {}, acc1 = {};
    // pipeline prologue: tile 0 in registers
    v8h av = load_a_slice(xh, ap, aoy, aox, akk, 0);
    v8h bv = *(const v8h*)(wrow + bkk);

    for (int k0 = 0; k0 < KTOT; k0 += 32) {
        *(v8h*)(As + apl * 32 + akk) = av;
        *(v8h*)(Bs + bco * 32 + bkk) = bv;
        __syncthreads();

        // issue next tile's global loads early (latency hidden behind WMMA phase)
        const int k1 = k0 + 32;
        if (k1 < KTOT) {
            __builtin_prefetch(wrow + k1 + 32, 0, 1);       // global_prefetch_b8
            av = load_a_slice(xh, ap, aoy, aox, akk, k1);
            bv = *(const v8h*)(wrow + k1 + bkk);
        }

        // fragments: contiguous 16B LDS reads
        union { v8h h[2]; v16h v; } ua, ub0, ub1;
        ua.h[0]  = *(const v8h*)(As + arow * 32 + akb);
        ua.h[1]  = *(const v8h*)(As + arow * 32 + akb + 8);
        ub0.h[0] = *(const v8h*)(Bs + bcol0 * 32 + bkb);
        ub0.h[1] = *(const v8h*)(Bs + bcol0 * 32 + bkb + 8);
        ub1.h[0] = *(const v8h*)(Bs + bcol1 * 32 + bkb);
        ub1.h[1] = *(const v8h*)(Bs + bcol1 * 32 + bkb + 8);
        acc0 = __builtin_amdgcn_wmma_f32_16x16x32_f16(false, ua.v, false, ub0.v,
                                                      (short)0, acc0, false, false);
        acc1 = __builtin_amdgcn_wmma_f32_16x16x32_f16(false, ua.v, false, ub1.v,
                                                      (short)0, acc1, false, false);
        __syncthreads();
    }

    // C/D: VGPR r -> M=(lane>=16)*8+r, N=lane&15; store h as f16 HWC
    const int co0 = coBase + wn * 32 + (lane & 15);
    const int co1 = co0 + 16;
    const float bias0 = b1[co0], bias1 = b1[co1];
    #pragma unroll
    for (int r = 0; r < 8; ++r) {
        int px = pxBase + wm * 16 + (lane >> 4) * 8 + r;
        float v0 = acc0[r] + bias0;
        float v1 = acc1[r] + bias1;
        h16[(size_t)px * COUT + co0] = (_Float16)(v0 > 0.f ? v0 : 0.f);
        h16[(size_t)px * COUT + co1] = (_Float16)(v1 > 0.f ? v1 : 0.f);
    }
}

// ---------------------------------------------------------------------------
// Kernel 2: fused 1x1 heads. One 16x16 tile per wave, fragments loaded
// directly from global with b128s (no LDS). N=64 (36 locs + 18 scores + pad).
// ---------------------------------------------------------------------------
__global__ void head1x1_wmma(const _Float16* __restrict__ h16, const _Float16* __restrict__ wh,
                             const float* __restrict__ bl, const float* __restrict__ bsc,
                             float* __restrict__ out)
{
    const int tid = threadIdx.x, lane = tid & 31, wave = tid >> 5;
    int tile = blockIdx.x * 8 + wave;           // wave-uniform
    if (tile >= 157 * 4) return;
    int mt = tile >> 2, nt = tile & 3;

    const int arow = mt * 16 + (lane & 15);     // <= 2511 < 2560 padded rows
    const int akb  = (lane >> 4) * 16;
    const int o    = nt * 16 + (lane & 15);
    const int bkb  = (lane >> 4) * 16;
    const _Float16* hrow = h16 + (size_t)arow * CIN;
    const _Float16* wrow = wh + (size_t)o * CIN;

    v8f acc = {};
    for (int k0 = 0; k0 < CIN; k0 += 32) {
        union { v8h h[2]; v16h v; } ua, ub;
        ua.h[0] = *(const v8h*)(hrow + k0 + akb);
        ua.h[1] = *(const v8h*)(hrow + k0 + akb + 8);
        ub.h[0] = *(const v8h*)(wrow + k0 + bkb);
        ub.h[1] = *(const v8h*)(wrow + k0 + bkb + 8);
        acc = __builtin_amdgcn_wmma_f32_16x16x32_f16(false, ua.v, false, ub.v,
                                                     (short)0, acc, false, false);
    }

    float bias = (o < 36) ? bl[o] : ((o < 54) ? bsc[o - 36] : 0.f);
    #pragma unroll
    for (int r = 0; r < 8; ++r) {
        int px = mt * 16 + (lane >> 4) * 8 + r;
        if (px < PIX) {
            float v = acc[r] + bias;
            if (o < 36)      out[OUT_LOCS   + (size_t)px * 36 + o] = v;
            else if (o < 54) out[OUT_SCORES + (size_t)px * 18 + (o - 36)] = v;
        }
    }
}

// ---------------------------------------------------------------------------
// Kernel 3: anchors + softmax(fg) + loc2bbox + clip + min-size mask.
// ---------------------------------------------------------------------------
__global__ void propose_prep(const float* __restrict__ out, float* __restrict__ anch,
                             float* __restrict__ roi, float* __restrict__ scoreM)
{
    int i = blockIdx.x * 256 + threadIdx.x;
    if (i >= NANCH) return;
    int p = i / 9, t = i - p * 9;
    int oy = p / WWs, ox = p - oy * WWs;

    const float ratios[3] = {0.5f, 1.f, 2.f};
    const float scales[3] = {8.f, 16.f, 32.f};
    float ratio = ratios[t / 3], scale = scales[t % 3];
    float ha = 16.f * scale * sqrtf(ratio);
    float wa = 16.f * scale * sqrtf(1.f / ratio);
    float cy = oy * 16.f + 8.f, cx = ox * 16.f + 8.f;
    float a0 = cy - 0.5f * ha, a1 = cx - 0.5f * wa;
    float a2 = cy + 0.5f * ha, a3 = cx + 0.5f * wa;
    anch[(size_t)i * 4 + 0] = a0; anch[(size_t)i * 4 + 1] = a1;
    anch[(size_t)i * 4 + 2] = a2; anch[(size_t)i * 4 + 3] = a3;

    float dy = out[OUT_LOCS + (size_t)i * 4 + 0];
    float dx = out[OUT_LOCS + (size_t)i * 4 + 1];
    float dh = out[OUT_LOCS + (size_t)i * 4 + 2];
    float dw = out[OUT_LOCS + (size_t)i * 4 + 3];
    float sh = a2 - a0, sw = a3 - a1;
    float ncy = dy * sh + (a0 + 0.5f * sh);
    float ncx = dx * sw + (a1 + 0.5f * sw);
    float nh  = expf(dh) * sh, nw = expf(dw) * sw;
    float y1 = fminf(fmaxf(ncy - 0.5f * nh, 0.f), 800.f);
    float x1 = fminf(fmaxf(ncx - 0.5f * nw, 0.f), 800.f);
    float y2 = fminf(fmaxf(ncy + 0.5f * nh, 0.f), 800.f);
    float x2 = fminf(fmaxf(ncx + 0.5f * nw, 0.f), 800.f);
    roi[(size_t)i * 4 + 0] = y1; roi[(size_t)i * 4 + 1] = x1;
    roi[(size_t)i * 4 + 2] = y2; roi[(size_t)i * 4 + 3] = x2;

    bool valid = ((y2 - y1) >= 16.f) && ((x2 - x1) >= 16.f);
    float s0 = out[OUT_SCORES + (size_t)p * 18 + 2 * t];
    float s1 = out[OUT_SCORES + (size_t)p * 18 + 2 * t + 1];
    float fg = 1.f / (1.f + expf(s0 - s1));
    scoreM[i] = valid ? fg : NEG_INF;
}

// ---------------------------------------------------------------------------
// Bitonic argsort (descending) over 32768 (score, index) pairs.
// ---------------------------------------------------------------------------
__global__ void sort_init(const float* __restrict__ scoreM,
                          float* __restrict__ key, int* __restrict__ idx)
{
    int i = blockIdx.x * 256 + threadIdx.x;
    key[i] = (i < NANCH) ? scoreM[i] : NEG_INF;
    idx[i] = i;
}

__global__ void bitonic_step(float* __restrict__ key, int* __restrict__ idx, int j, int k)
{
    int i = blockIdx.x * 256 + threadIdx.x;
    int ixj = i ^ j;
    if (ixj > i) {
        float ki = key[i], kj = key[ixj];
        bool up = ((i & k) == 0);
        bool doSwap = up ? (ki < kj) : (ki > kj);   // descending
        if (doSwap) {
            key[i] = kj; key[ixj] = ki;
            int ti = idx[i]; idx[i] = idx[ixj]; idx[ixj] = ti;
        }
    }
}

__global__ void gather_sorted(const float* __restrict__ key, const int* __restrict__ idx,
                              const float* __restrict__ roi,
                              float* __restrict__ boxS, int* __restrict__ valS)
{
    int i = blockIdx.x * 256 + threadIdx.x;
    if (i >= PPRE) return;
    int j = idx[i];
    boxS[(size_t)i * 4 + 0] = roi[(size_t)j * 4 + 0];
    boxS[(size_t)i * 4 + 1] = roi[(size_t)j * 4 + 1];
    boxS[(size_t)i * 4 + 2] = roi[(size_t)j * 4 + 2];
    boxS[(size_t)i * 4 + 3] = roi[(size_t)j * 4 + 3];
    valS[i] = (key[i] > -3.0e38f) ? 1 : 0;
}

// ---------------------------------------------------------------------------
// NMS bitmask (64x64 tiles): bit j set iff j>i & IoU>0.7.
// ---------------------------------------------------------------------------
__global__ void nms_mask(const float* __restrict__ boxes, unsigned long long* __restrict__ mask)
{
    __shared__ float cb[64 * 4];
    int br = blockIdx.y, bc = blockIdx.x, t = threadIdx.x;
    int cj = bc * 64 + t;
    if (cj < PPRE) {
        cb[t * 4 + 0] = boxes[(size_t)cj * 4 + 0];
        cb[t * 4 + 1] = boxes[(size_t)cj * 4 + 1];
        cb[t * 4 + 2] = boxes[(size_t)cj * 4 + 2];
        cb[t * 4 + 3] = boxes[(size_t)cj * 4 + 3];
    }
    __syncthreads();
    int i = br * 64 + t;
    if (i >= PPRE) return;
    float y1 = boxes[(size_t)i * 4 + 0], x1 = boxes[(size_t)i * 4 + 1];
    float y2 = boxes[(size_t)i * 4 + 2], x2 = boxes[(size_t)i * 4 + 3];
    float area = (y2 - y1) * (x2 - x1);
    unsigned long long bits = 0ull;
    int lim = PPRE - bc * 64; if (lim > 64) lim = 64;
    for (int jj = 0; jj < lim; ++jj) {
        int j = bc * 64 + jj;
        if (j <= i) continue;
        float by1 = cb[jj * 4 + 0], bx1 = cb[jj * 4 + 1];
        float by2 = cb[jj * 4 + 2], bx2 = cb[jj * 4 + 3];
        float ih = fminf(y2, by2) - fmaxf(y1, by1); ih = ih > 0.f ? ih : 0.f;
        float iw = fminf(x2, bx2) - fmaxf(x1, bx1); iw = iw > 0.f ? iw : 0.f;
        float inter = ih * iw;
        float iou = inter / (area + (by2 - by1) * (bx2 - bx1) - inter + 1e-12f);
        if (iou > 0.7f) bits |= (1ull << jj);
    }
    mask[(size_t)i * NBW + bc] = bits;
}

// Serial greedy sweep (matches reference fori_loop), compacts kept boxes.
__global__ void nms_sweep(const float* __restrict__ boxes, const int* __restrict__ valid,
                          const unsigned long long* __restrict__ mask,
                          float* __restrict__ rois, int* __restrict__ roiIdx)
{
    __shared__ unsigned long long rem[NBW];
    __shared__ int keepFlag;
    __shared__ int cnt;
    int t = threadIdx.x;
    for (int w = t; w < NBW; w += 256) rem[w] = 0ull;
    if (t == 0) cnt = 0;
    __syncthreads();
    for (int i = 0; i < PPRE; ++i) {
        if (t == 0) {
            bool sup = (rem[i >> 6] >> (i & 63)) & 1ull;
            keepFlag = (valid[i] && !sup) ? 1 : 0;
        }
        __syncthreads();
        if (keepFlag) {
            for (int w = t; w < NBW; w += 256) rem[w] |= mask[(size_t)i * NBW + w];
            if (t == 0) {
                if (cnt < NPOST) {
                    rois[(size_t)cnt * 4 + 0] = boxes[(size_t)i * 4 + 0];
                    rois[(size_t)cnt * 4 + 1] = boxes[(size_t)i * 4 + 1];
                    rois[(size_t)cnt * 4 + 2] = boxes[(size_t)i * 4 + 2];
                    rois[(size_t)cnt * 4 + 3] = boxes[(size_t)i * 4 + 3];
                }
                cnt++;
            }
        }
        __syncthreads();
    }
    int c = cnt;
    for (int r = t; r < NPOST; r += 256) {
        if (r >= c) {
            rois[(size_t)r * 4 + 0] = 0.f; rois[(size_t)r * 4 + 1] = 0.f;
            rois[(size_t)r * 4 + 2] = 0.f; rois[(size_t)r * 4 + 3] = 0.f;
        }
        roiIdx[r] = 0;
    }
}

// ---------------------------------------------------------------------------
extern "C" void kernel_launch(void* const* d_in, const int* in_sizes, int n_in,
                              void* d_out, int out_size, void* d_ws, size_t ws_size,
                              hipStream_t stream)
{
    const float* x   = (const float*)d_in[0];
    const float* W1  = (const float*)d_in[1];
    const float* b1  = (const float*)d_in[2];
    const float* Wsc = (const float*)d_in[3];
    const float* bsc = (const float*)d_in[4];
    const float* Wl  = (const float*)d_in[5];
    const float* bl  = (const float*)d_in[6];

    float* out = (float*)d_out;
    char*  ws  = (char*)d_ws;
    _Float16* xh    = (_Float16*)(ws + WS_XH);
    _Float16* wp    = (_Float16*)(ws + WS_WPACK);
    _Float16* h16   = (_Float16*)(ws + WS_H16);
    _Float16* wh    = (_Float16*)(ws + WS_WHEAD);
    float* roi      = (float*)(ws + WS_ROI);
    float* scoreM   = (float*)(ws + WS_SCOREM);
    float* key      = (float*)(ws + WS_KEY);
    int*   idx      = (int*)  (ws + WS_IDX);
    float* boxS     = (float*)(ws + WS_BOXS);
    int*   valS     = (int*)  (ws + WS_VALS);
    unsigned long long* mask = (unsigned long long*)(ws + WS_MASK);

    // 0) layout packs
    pack_x_hwc<<<dim3(79, 16), dim3(32, 8), 0, stream>>>(x, xh);
    pack_w1<<<(COUT * KTOT + 255) / 256, 256, 0, stream>>>(W1, wp);
    pack_whead<<<(64 * CIN + 255) / 256, 256, 0, stream>>>(Wl, Wsc, wh);
    // 1) 3x3 conv + ReLU (8 cout-tiles of 64 x 40 pixel-tiles of 64)
    conv3x3_wmma<<<dim3(8, 40), 256, 0, stream>>>(xh, wp, b1, h16);
    // 2) fused 1x1 heads -> locs / raw scores into d_out
    head1x1_wmma<<<79, 256, 0, stream>>>(h16, wh, bl, bsc, out);
    // 3) anchors + fg score + loc2bbox + clip + mask
    propose_prep<<<(NANCH + 255) / 256, 256, 0, stream>>>(out, out + OUT_ANCH, roi, scoreM);
    // 4) descending argsort (bitonic, 32768 pad)
    sort_init<<<NSORT / 256, 256, 0, stream>>>(scoreM, key, idx);
    for (int k = 2; k <= NSORT; k <<= 1)
        for (int j = k >> 1; j > 0; j >>= 1)
            bitonic_step<<<NSORT / 256, 256, 0, stream>>>(key, idx, j, k);
    // 5) gather top-12000 + validity
    gather_sorted<<<(PPRE + 255) / 256, 256, 0, stream>>>(key, idx, roi, boxS, valS);
    // 6) NMS bitmask + serial greedy sweep + compaction
    nms_mask<<<dim3(NBW, NBW), 64, 0, stream>>>(boxS, mask);
    nms_sweep<<<1, 256, 0, stream>>>(boxS, valS, mask,
                                     out + OUT_ROIS, (int*)(out + OUT_RIDX));
}
